// UserAttention_30296699306117
// MI455X (gfx1250) — compile-verified, matrix operand order checked by reading, not run on hardware
//
#include <hip/hip_runtime.h>
#include <hip/hip_bf16.h>
#include <math.h>

// ---------------------------------------------------------------------------
// Problem constants (from the reference): B=2048, T=200, C=128
// ---------------------------------------------------------------------------
#define CC 128
#define TT 200
#define QTILE 16          // batch rows per block in the WMMA query kernel
#define UQ_STRIDE 132     // padded LDS stride (float4-aligned, low bank conflicts)
#define KSTR 132          // padded LDS stride for keys tile

typedef __attribute__((ext_vector_type(2)))  float    v2f;
typedef __attribute__((ext_vector_type(8)))  float    v8f;
typedef __attribute__((ext_vector_type(16))) _Float16 v16h;

#if __has_builtin(__builtin_amdgcn_wmma_f32_16x16x4_f32)
#define USE_F32_WMMA 1
#endif

// ---------------------------------------------------------------------------
// Kernel 1: Q = tanh(UQ @ W + b)   -- (B x 128) * (128 x 128) GEMM via WMMA.
// One block = 256 threads = 8 waves. Block handles a 16-batch tile; wave w
// produces the 16x16 output tile at columns [16w, 16w+16).
// ---------------------------------------------------------------------------
__global__ __launch_bounds__(256)
void query_tanh_wmma(const float* __restrict__ uq,    // [B, C]
                     const float* __restrict__ W,     // [C, C] row-major (c -> d)
                     const float* __restrict__ bias,  // [C]
                     float* __restrict__ qout,        // [B, C]
                     int B)
{
    __shared__ float uq_lds[QTILE * UQ_STRIDE];

    const int tid = threadIdx.x;
    const int bt0 = blockIdx.x * QTILE;

    // Stage the 16x128 A tile (float4 loads, padded LDS stride)
    for (int i = tid; i < QTILE * (CC / 4); i += 256) {
        const int r  = i / (CC / 4);
        const int c4 = i % (CC / 4);
        float4 v = make_float4(0.f, 0.f, 0.f, 0.f);
        if (bt0 + r < B)
            v = *(const float4*)(uq + (size_t)(bt0 + r) * CC + c4 * 4);
        *(float4*)(&uq_lds[r * UQ_STRIDE + c4 * 4]) = v;
    }
    __syncthreads();

    const int wave = tid >> 5;       // 0..7
    const int lane = tid & 31;
    const int n0   = wave * 16;      // output-column tile
    const int hi   = lane >> 4;      // 0 or 1 (upper half of wave)
    const int lrow = lane & 15;

    v8f acc = {};

#if defined(USE_F32_WMMA)
    // V_WMMA_F32_16X16X4_F32: A 16x4, B 4x16, 32 K-steps of 4.
    // 32-bit A layout: lanes 0-15 hold K=0 (v0) / K=1 (v1); lanes 16-31 K=2/K=3.
    for (int k = 0; k < CC; k += 4) {
        v2f a, bf;
        a.x  = uq_lds[lrow * UQ_STRIDE + k + 2 * hi + 0];
        a.y  = uq_lds[lrow * UQ_STRIDE + k + 2 * hi + 1];
        bf.x = W[(size_t)(k + 2 * hi + 0) * CC + n0 + lrow];
        bf.y = W[(size_t)(k + 2 * hi + 1) * CC + n0 + lrow];
        acc = __builtin_amdgcn_wmma_f32_16x16x4_f32(
            /*neg_a=*/false, a, /*neg_b=*/false, bf,
            /*c_mod=*/(short)0, acc, /*reuse_a=*/false, /*reuse_b=*/false);
    }
#else
    // Fallback: codegen-confirmed V_WMMA_F32_16X16X32_F16 (4 K-steps of 32).
    // 16-bit A 16x32 layout: lanes 0-15: elems 0-7 -> K=k0+0..7, elems 8-15 ->
    // K=k0+16..23; lanes 16-31: +8. B (32x16) mirrored.
    for (int k0 = 0; k0 < CC; k0 += 32) {
        v16h a, bf;
#pragma unroll
        for (int j = 0; j < 8; ++j) {
            a[j]      = (_Float16)uq_lds[lrow * UQ_STRIDE + k0 + hi * 8 + j];
            a[8 + j]  = (_Float16)uq_lds[lrow * UQ_STRIDE + k0 + 16 + hi * 8 + j];
            bf[j]     = (_Float16)W[(size_t)(k0 + hi * 8 + j) * CC + n0 + lrow];
            bf[8 + j] = (_Float16)W[(size_t)(k0 + 16 + hi * 8 + j) * CC + n0 + lrow];
        }
        acc = __builtin_amdgcn_wmma_f32_16x16x32_f16(
            false, a, false, bf, (short)0, acc, false, false);
    }
#endif

    // D layout: VGPR r -> M = r + 8*hi, N = lrow. Apply bias + tanh, store.
    const float bv = bias[n0 + lrow];
#pragma unroll
    for (int r = 0; r < 8; ++r) {
        const int m = r + 8 * hi;
        if (bt0 + m < B)
            qout[(size_t)(bt0 + m) * CC + n0 + lrow] = tanhf(acc[r] + bv);
    }
}

// ---------------------------------------------------------------------------
// Kernel 2: per-batch masked softmax attention + residual-mean, fully fused.
// One block (256 threads) per batch. Stage keys[b] (200x128 f32, 103 KB) in
// LDS once; compute align (dot), masked softmax, weighted sum, + mean(keys).
// ---------------------------------------------------------------------------
__global__ __launch_bounds__(256)
void attn_softmax_fused(const float* __restrict__ keys,   // [B, T, C]
                        const int*   __restrict__ klen,   // [B]
                        const float* __restrict__ q,      // [B, C]
                        float* __restrict__ out)          // [B, C]
{
    __shared__ float k_lds[TT * KSTR];
    __shared__ float q_lds[CC];
    __shared__ float a_lds[256];
    __shared__ float red[256];
    __shared__ int   s_len;

    const int b   = blockIdx.x;
    const int tid = threadIdx.x;

    if (tid < CC) q_lds[tid] = q[(size_t)b * CC + tid];
    if (tid == 0) s_len = klen[b];

    // Stage full keys tile (6400 float4 loads across 256 threads)
    const float* kb = keys + (size_t)b * TT * CC;
    for (int i = tid; i < TT * (CC / 4); i += 256) {
        const int t  = i / (CC / 4);
        const int c4 = i % (CC / 4);
        const float4 v = *(const float4*)(kb + (size_t)t * CC + c4 * 4);
        *(float4*)(&k_lds[t * KSTR + c4 * 4]) = v;
    }
    __syncthreads();

    const int   L     = s_len;
    const float SCALE = 0.08838834764831845f;   // 1/sqrt(128)
    const float PADV  = -4294967295.0f;         // -(2^32) + 1

    // align[t] = (q . keys[t]) * scale, masked
    float aval;
    if (tid < TT) {
        float dot = 0.f;
        const float* kr = &k_lds[tid * KSTR];
#pragma unroll 8
        for (int c = 0; c < CC; ++c) dot = fmaf(q_lds[c], kr[c], dot);
        aval = (tid < L) ? dot * SCALE : PADV;
    } else {
        aval = -INFINITY;                        // never wins max; exp -> 0
    }
    a_lds[tid] = aval;
    red[tid]   = aval;
    __syncthreads();

    // max reduction
    for (int s = 128; s > 0; s >>= 1) {
        if (tid < s) red[tid] = fmaxf(red[tid], red[tid + s]);
        __syncthreads();
    }
    const float mx = red[0];
    __syncthreads();

    // exp + sum reduction
    const float e = (tid < TT) ? __expf(a_lds[tid] - mx) : 0.f;
    a_lds[tid] = e;
    red[tid]   = e;
    __syncthreads();
    for (int s = 128; s > 0; s >>= 1) {
        if (tid < s) red[tid] += red[tid + s];
        __syncthreads();
    }
    const float inv = 1.0f / red[0];

    // out[c] = sum_t w[t]*keys[t,c] + mean_t keys[t,c]
    if (tid < CC) {
        float acc = 0.f, msum = 0.f;
        for (int t = 0; t < TT; ++t) {
            const float kv = k_lds[t * KSTR + tid];
            acc  = fmaf(a_lds[t], kv, acc);
            msum += kv;
        }
        out[(size_t)b * CC + tid] = acc * inv + msum * (1.0f / TT);
    }
}

// ---------------------------------------------------------------------------
// Launch: inputs are (user_query, keys, keys_length, W, b) in dict order.
// ---------------------------------------------------------------------------
extern "C" void kernel_launch(void* const* d_in, const int* in_sizes, int n_in,
                              void* d_out, int out_size, void* d_ws, size_t ws_size,
                              hipStream_t stream) {
    const float* uq   = (const float*)d_in[0];   // [B,1,C]
    const float* keys = (const float*)d_in[1];   // [B,T,C]
    const int*   klen = (const int*)d_in[2];     // [B,1]
    const float* W    = (const float*)d_in[3];   // [C,C]
    const float* bias = (const float*)d_in[4];   // [C]

    const int B = in_sizes[0] / CC;

    float* q_ws = (float*)d_ws;                  // [B, C] scratch (1 MB for B=2048)

    const int qblocks = (B + QTILE - 1) / QTILE;
    query_tanh_wmma<<<qblocks, 256, 0, stream>>>(uq, W, bias, q_ws, B);
    attn_softmax_fused<<<B, 256, 0, stream>>>(keys, klen, q_ws, (float*)d_out);
}
